// GNNStack_45337674777304
// MI455X (gfx1250) — matrix operand order repaired and checked by reading, not compiled, not run on hardware
//
#include <hip/hip_runtime.h>
#include <hip/hip_bf16.h>

typedef __attribute__((ext_vector_type(16))) _Float16 v16h;
typedef __attribute__((ext_vector_type(8)))  float    v8f;

#define N_NODES 100000
#define N_EDGES 1000000
#define HID 128
#define OUT_DIM 40

// ---------------------------------------------------------------------------
// Weight conversion f32 -> f16 (Wp2 padded from 40 to 48 rows with zeros)
// ---------------------------------------------------------------------------
__global__ void cvt_weights_kernel(const float* __restrict__ W1,
                                   const float* __restrict__ W2,
                                   const float* __restrict__ Wp1,
                                   const float* __restrict__ Wp2,
                                   _Float16* __restrict__ W1h,
                                   _Float16* __restrict__ W2h,
                                   _Float16* __restrict__ Wp1h,
                                   _Float16* __restrict__ Wp2h) {
    int i = blockIdx.x * blockDim.x + threadIdx.x;
    if (i < 16384) {
        W1h[i] = (_Float16)W1[i];
    } else if (i < 32768) {
        W2h[i - 16384] = (_Float16)W2[i - 16384];
    } else if (i < 49152) {
        Wp1h[i - 32768] = (_Float16)Wp1[i - 32768];
    } else if (i < 55296) {
        int j = i - 49152;                       // 48 x 128 padded
        Wp2h[j] = (j < OUT_DIM * HID) ? (_Float16)Wp2[j] : (_Float16)0.f;
    }
}

// ---------------------------------------------------------------------------
// WMMA GEMM:  C[M x nmax] = A[M x 128](f32) * Wh[N x 128](f16)^T + bias
// One wave computes a 16(M) x (NT*16)(N) tile via v_wmma_f32_16x16x32_f16.
// A fragment: lanes 0-15: K {0..15}, lanes 16-31: K {16..31}
//             (contiguous, matching B), M = lane&15.
// B fragment: column n = lane&15 holds K contiguous (half*16 + 0..15);
//             since C = X*W^T, column n of B == row n of W (contiguous).
// ---------------------------------------------------------------------------
template <int NT>
__global__ void __launch_bounds__(256)
gemm_wmma_kernel(const float* __restrict__ A, const _Float16* __restrict__ Wh,
                 const float* __restrict__ bias, float* __restrict__ C,
                 int M, int ldc, int nmax) {
    const int lane = threadIdx.x & 31;
    const int wave = threadIdx.x >> 5;
    const int half = lane >> 4;      // 0 or 1
    const int l15  = lane & 15;
    const int row0 = blockIdx.x * 128 + wave * 16;

    int arow = row0 + l15;
    if (arow >= M) arow = M - 1;     // clamp loads; stores are guarded
    const float* __restrict__ arowp = A + (size_t)arow * HID;

    v8f acc[NT] = {};

#pragma unroll
    for (int k0 = 0; k0 < HID; k0 += 32) {
        // ---- A fragment (f32 -> f16 convert in-register) ----
        const float* ap = arowp + k0 + half * 16;
        v16h a;
#pragma unroll
        for (int i = 0; i < 8; ++i) {
            a[i]     = (_Float16)ap[i];
            a[8 + i] = (_Float16)ap[8 + i];
        }
        // ---- B fragments + WMMA ----
#pragma unroll
        for (int t = 0; t < NT; ++t) {
            const _Float16* bp = Wh + (size_t)(t * 16 + l15) * HID + k0 + half * 16;
            v16h b = *(const v16h*)bp;   // 16 contiguous f16 (32B)
            acc[t] = __builtin_amdgcn_wmma_f32_16x16x32_f16(
                false, a, false, b, (short)0, acc[t], false, false);
        }
    }

    // ---- store: 32-bit C/D layout: VGPR r -> (m = r + half*8, n = l15) ----
#pragma unroll
    for (int t = 0; t < NT; ++t) {
        int n = t * 16 + l15;
        if (n < nmax) {
            float bv = bias[n];
#pragma unroll
            for (int r = 0; r < 8; ++r) {
                int m = row0 + half * 8 + r;
                if (m < M) C[(size_t)m * ldc + n] = acc[t][r] + bv;
            }
        }
    }
}

// ---------------------------------------------------------------------------
// Ordered-uint mapping for float atomic max
// ---------------------------------------------------------------------------
__device__ __forceinline__ unsigned mapf(float f) {
    unsigned u = __float_as_uint(f);
    return (u & 0x80000000u) ? ~u : (u | 0x80000000u);
}
__device__ __forceinline__ float unmapf(unsigned u) {
    return __uint_as_float((u & 0x80000000u) ? (u & 0x7fffffffu) : ~u);
}

// ---------------------------------------------------------------------------
// Per-node attention scores (wave per node) + init of m / denom / agg
// sdst[n] = <h[n], att[0:128]>   ssrc[n] = <h[n], att[128:256]>
// ---------------------------------------------------------------------------
__global__ void __launch_bounds__(256)
scores_init_kernel(const float* __restrict__ h, const float* __restrict__ att,
                   float* __restrict__ sdst, float* __restrict__ ssrc,
                   unsigned* __restrict__ mmax, float* __restrict__ denom,
                   float* __restrict__ agg, int M) {
    int gw = (int)((blockIdx.x * blockDim.x + threadIdx.x) >> 5);
    if (gw >= M) return;
    int lane = threadIdx.x & 31;

    float4 hv = *(const float4*)(h + (size_t)gw * HID + lane * 4);
    float4 ai = *(const float4*)(att + lane * 4);
    float4 aj = *(const float4*)(att + HID + lane * 4);
    float si = hv.x * ai.x + hv.y * ai.y + hv.z * ai.z + hv.w * ai.w;
    float sj = hv.x * aj.x + hv.y * aj.y + hv.z * aj.z + hv.w * aj.w;
#pragma unroll
    for (int off = 16; off > 0; off >>= 1) {
        si += __shfl_xor(si, off);
        sj += __shfl_xor(sj, off);
    }
    *(float4*)(agg + (size_t)gw * HID + lane * 4) = make_float4(0.f, 0.f, 0.f, 0.f);
    if (lane == 0) {
        sdst[gw]  = si;
        ssrc[gw]  = sj;
        mmax[gw]  = 0x007fffffu;   // mapf(-inf)
        denom[gw] = 0.f;
    }
}

// ---------------------------------------------------------------------------
// Edge pass 1: e = leaky(sdst[dst] + ssrc[src]); store; segment-max into mmax
// ---------------------------------------------------------------------------
__global__ void __launch_bounds__(256)
edge_pass1_kernel(const long long* __restrict__ ei,
                  const float* __restrict__ sdst, const float* __restrict__ ssrc,
                  float* __restrict__ ebuf, unsigned* __restrict__ mmax) {
    int i = blockIdx.x * blockDim.x + threadIdx.x;
    if (i >= N_EDGES) return;
    long long s = ei[i];
    long long d = ei[N_EDGES + i];
    float e = sdst[d] + ssrc[s];
    e = e > 0.f ? e : 0.2f * e;
    ebuf[i] = e;
    atomicMax(&mmax[d], mapf(e));
}

// ---------------------------------------------------------------------------
// Edge pass 2: ex = exp(e - m[dst]); store; segment-sum into denom
// ---------------------------------------------------------------------------
__global__ void __launch_bounds__(256)
edge_pass2_kernel(const long long* __restrict__ ei,
                  const unsigned* __restrict__ mmax,
                  float* __restrict__ ebuf, float* __restrict__ denom) {
    int i = blockIdx.x * blockDim.x + threadIdx.x;
    if (i >= N_EDGES) return;
    long long d = ei[N_EDGES + i];
    float ex = __expf(ebuf[i] - unmapf(mmax[d]));
    ebuf[i] = ex;
    atomicAdd(&denom[d], ex);
}

// ---------------------------------------------------------------------------
// Edge pass 3 (wave per edge): agg[dst] += h[src] * (ex / denom[dst])
// ---------------------------------------------------------------------------
__global__ void __launch_bounds__(256)
edge_pass3_kernel(const long long* __restrict__ ei,
                  const float* __restrict__ ebuf, const float* __restrict__ denom,
                  const float* __restrict__ h, float* __restrict__ agg) {
    int gw = (int)((blockIdx.x * blockDim.x + threadIdx.x) >> 5);
    if (gw >= N_EDGES) return;
    int lane = threadIdx.x & 31;
    long long s = ei[gw];
    long long d = ei[N_EDGES + gw];
    float alpha = ebuf[gw] / denom[d];
    float4 hv = *(const float4*)(h + (size_t)s * HID + lane * 4);
    float* ao = agg + (size_t)d * HID + lane * 4;
    atomicAdd(ao + 0, hv.x * alpha);
    atomicAdd(ao + 1, hv.y * alpha);
    atomicAdd(ao + 2, hv.z * alpha);
    atomicAdd(ao + 3, hv.w * alpha);
}

// ---------------------------------------------------------------------------
// out = relu(in + biasvec), float4 per thread
// ---------------------------------------------------------------------------
__global__ void __launch_bounds__(256)
bias_relu_kernel(const float* __restrict__ in, const float* __restrict__ bias,
                 float* __restrict__ out, int M) {
    int t = blockIdx.x * blockDim.x + threadIdx.x;
    if (t >= M * (HID / 4)) return;
    int c4 = (t & 31) * 4;
    float4 v = *(const float4*)(in + (size_t)t * 4);
    float4 b = *(const float4*)(bias + c4);
    v.x = fmaxf(v.x + b.x, 0.f);
    v.y = fmaxf(v.y + b.y, 0.f);
    v.z = fmaxf(v.z + b.z, 0.f);
    v.w = fmaxf(v.w + b.w, 0.f);
    *(float4*)(out + (size_t)t * 4) = v;
}

// ---------------------------------------------------------------------------
// log_softmax over 40 classes, wave per node
// ---------------------------------------------------------------------------
__global__ void __launch_bounds__(256)
log_softmax_kernel(const float* __restrict__ y, float* __restrict__ out, int M) {
    int gw = (int)((blockIdx.x * blockDim.x + threadIdx.x) >> 5);
    if (gw >= M) return;
    int lane = threadIdx.x & 31;
    const float* yr = y + (size_t)gw * OUT_DIM;
    float v0 = yr[lane];
    float v1 = (lane < OUT_DIM - 32) ? yr[32 + lane] : -3.402823466e38f;
    float mx = fmaxf(v0, v1);
#pragma unroll
    for (int off = 16; off > 0; off >>= 1) mx = fmaxf(mx, __shfl_xor(mx, off));
    float sum = __expf(v0 - mx) + ((lane < OUT_DIM - 32) ? __expf(v1 - mx) : 0.f);
#pragma unroll
    for (int off = 16; off > 0; off >>= 1) sum += __shfl_xor(sum, off);
    float lse = mx + __logf(sum);
    float* orow = out + (size_t)gw * OUT_DIM;
    orow[lane] = v0 - lse;
    if (lane < OUT_DIM - 32) orow[32 + lane] = v1 - lse;
}

// ---------------------------------------------------------------------------
// Launch
// ---------------------------------------------------------------------------
extern "C" void kernel_launch(void* const* d_in, const int* in_sizes, int n_in,
                              void* d_out, int out_size, void* d_ws, size_t ws_size,
                              hipStream_t stream) {
    const float*     x     = (const float*)d_in[0];
    const long long* ei    = (const long long*)d_in[1];
    const float*     W1    = (const float*)d_in[2];
    const float*     b1    = (const float*)d_in[3];
    const float*     att1  = (const float*)d_in[4];
    const float*     bias1 = (const float*)d_in[5];
    const float*     W2    = (const float*)d_in[6];
    const float*     b2    = (const float*)d_in[7];
    const float*     att2  = (const float*)d_in[8];
    const float*     bias2 = (const float*)d_in[9];
    const float*     Wp1   = (const float*)d_in[10];
    const float*     bp1   = (const float*)d_in[11];
    const float*     Wp2   = (const float*)d_in[12];
    const float*     bp2   = (const float*)d_in[13];
    float*           out   = (float*)d_out;

    char* ws = (char*)d_ws;
    const size_t BUF = (size_t)N_NODES * HID * sizeof(float);   // 51,200,000
    float*    bufA  = (float*)(ws);
    float*    bufB  = (float*)(ws + BUF);
    char*     sc    = ws + 2 * BUF;
    float*    ebuf  = (float*)(sc);                       // 4,000,000 B
    float*    sdst  = (float*)(sc + 4000000);
    float*    ssrc  = (float*)(sc + 4400000);
    unsigned* mmax  = (unsigned*)(sc + 4800000);
    float*    denom = (float*)(sc + 5200000);
    _Float16* W1h   = (_Float16*)(sc + 5600000);
    _Float16* W2h   = W1h + 16384;
    _Float16* Wp1h  = W2h + 16384;
    _Float16* Wp2h  = Wp1h + 16384;     // 48 x 128 padded

    const int M = N_NODES;
    dim3 blk(256);
    dim3 gGemm((M + 127) / 128);
    dim3 gNodeWave(M / 8);              // 100000 waves exactly, 8 per block
    dim3 gEdge((N_EDGES + 255) / 256);
    dim3 gEdgeWave(N_EDGES / 8);        // wave per edge
    dim3 gElem((M * (HID / 4) + 255) / 256);

    // weights -> f16
    cvt_weights_kernel<<<dim3(216), blk, 0, stream>>>(W1, W2, Wp1, Wp2, W1h, W2h, Wp1h, Wp2h);

    // ---- layer 1 ----
    gemm_wmma_kernel<8><<<gGemm, blk, 0, stream>>>(x, W1h, b1, bufA, M, HID, HID);
    scores_init_kernel<<<gNodeWave, blk, 0, stream>>>(bufA, att1, sdst, ssrc, mmax, denom, bufB, M);
    edge_pass1_kernel<<<gEdge, blk, 0, stream>>>(ei, sdst, ssrc, ebuf, mmax);
    edge_pass2_kernel<<<gEdge, blk, 0, stream>>>(ei, mmax, ebuf, denom);
    edge_pass3_kernel<<<gEdgeWave, blk, 0, stream>>>(ei, ebuf, denom, bufA, bufB);
    bias_relu_kernel<<<gElem, blk, 0, stream>>>(bufB, bias1, bufA, M);   // hin2 -> bufA

    // ---- layer 2 ----
    gemm_wmma_kernel<8><<<gGemm, blk, 0, stream>>>(bufA, W2h, b2, bufB, M, HID, HID);
    scores_init_kernel<<<gNodeWave, blk, 0, stream>>>(bufB, att2, sdst, ssrc, mmax, denom, bufA, M);
    edge_pass1_kernel<<<gEdge, blk, 0, stream>>>(ei, sdst, ssrc, ebuf, mmax);
    edge_pass2_kernel<<<gEdge, blk, 0, stream>>>(ei, mmax, ebuf, denom);
    edge_pass3_kernel<<<gEdgeWave, blk, 0, stream>>>(ei, ebuf, denom, bufB, bufA);
    bias_relu_kernel<<<gElem, blk, 0, stream>>>(bufA, bias2, bufB, M);   // hfinal -> bufB

    // ---- MLP head + log_softmax ----
    gemm_wmma_kernel<8><<<gGemm, blk, 0, stream>>>(bufB, Wp1h, bp1, bufA, M, HID, HID);   // y1 -> bufA
    gemm_wmma_kernel<3><<<gGemm, blk, 0, stream>>>(bufA, Wp2h, bp2, bufB, M, OUT_DIM, OUT_DIM); // y -> bufB
    log_softmax_kernel<<<gNodeWave, blk, 0, stream>>>(bufB, out, M);
}
